// GNNModel_71768903516467
// MI455X (gfx1250) — matrix operand order repaired and checked by reading, not compiled, not run on hardware
//
#include <hip/hip_runtime.h>
#include <cstdint>

#define N_NODES  50000
#define N_EDGES  800000
#define CDIM     128
#define BN_EPS   1e-5f

// LDS row pitch for staged weights: 136 bf16 = 272 B = 68 dwords.
// Lane stride 68 dwords mod 64 banks = 4 -> the 16 lanes of a B-fragment
// hit disjoint 4-dword bank groups (conflict-free ds_load_b128 per half-wave).
#define WPITCH   136

typedef __bf16 bf16;
typedef __attribute__((ext_vector_type(16))) __bf16 v16bf;
typedef __attribute__((ext_vector_type(8)))  __bf16 v8bf;
typedef __attribute__((ext_vector_type(8)))  float  v8f;

// ---------------------------------------------------------------------------
// Weight prep: Wt[n][k] = bf16(W[k][n])  (col-major bf16, unpadded in global)
// ---------------------------------------------------------------------------
__global__ void prep_w_kernel(const float* __restrict__ W, bf16* __restrict__ Wt) {
    const int n = blockIdx.x;
    const int k = threadIdx.x;
    Wt[n * CDIM + k] = (bf16)W[k * CDIM + n];
}

// ---------------------------------------------------------------------------
// GEMM: out[M x 128] = A[M x 128] @ W[128 x 128] + bias, bf16 WMMA, f32 acc.
// Block = 128 threads (4 waves); wave w computes rows [blk*64 + w*16, +16).
// Weights staged once per block in LDS (34 KB of the 320 KB/WGP), padded
// pitch for bank-conflict-free ds_load_b128 B-fragment reads.
// Fragment layouts per CDNA5 ISA 7.12.2:
//   A 16x32 bf16 : lane l: row = l&15, kHalf = l>>4;
//                  elem i in [0,8):  K = k0 + kHalf*8 + i
//                  elem i in [8,16): K = k0 + 16 + kHalf*8 + (i-8)
//   B 32x16 bf16 : lane l: col = l&15, kHalf = l>>4; elem i: K same as A's elem i
//   C/D 16x16 f32: lane l: col = l&15; vgpr r: row = r + (l>=16 ? 8 : 0)
// ---------------------------------------------------------------------------
__global__ void __launch_bounds__(128)
gemm_bias_kernel(const float* __restrict__ A, const bf16* __restrict__ Wt,
                 const float* __restrict__ bias, float* __restrict__ out,
                 float* __restrict__ out2, int M) {
    __shared__ bf16 sW[CDIM * WPITCH];             // 34,816 B

    // --- cooperative stage of the 128x128 bf16 weight matrix into LDS
    for (int i = threadIdx.x; i < CDIM * (CDIM / 8); i += 128) {
        const int r = i >> 4;                      // weight row (output col n)
        const int c = (i & 15) * 8;                // 8 bf16 per 16B chunk
        *(uint4*)(sW + r * WPITCH + c) =
            ((const uint4*)(Wt + (long)r * CDIM))[i & 15];
    }
    __syncthreads();

    const int wave  = threadIdx.x >> 5;
    const int lane  = threadIdx.x & 31;
    const int l16   = lane & 15;
    const int kHalf = lane >> 4;
    const int row0  = blockIdx.x * 64 + wave * 16;

    int arow = row0 + l16;
    if (arow >= M) arow = M - 1;                   // clamp reads; stores guarded
    const float* ap = A + (long)arow * CDIM;

    v8f acc[8];
#pragma unroll
    for (int n = 0; n < 8; ++n)
        acc[n] = (v8f){0.f, 0.f, 0.f, 0.f, 0.f, 0.f, 0.f, 0.f};

#pragma unroll
    for (int k0 = 0; k0 < CDIM; k0 += 32) {
        // ---- A fragment: two runs of 8 consecutive f32, convert to bf16
        const float4 f0 = *(const float4*)(ap + k0 + kHalf * 8);
        const float4 f1 = *(const float4*)(ap + k0 + kHalf * 8 + 4);
        const float4 f2 = *(const float4*)(ap + k0 + 16 + kHalf * 8);
        const float4 f3 = *(const float4*)(ap + k0 + 16 + kHalf * 8 + 4);
        v16bf a;
        a[0]  = (bf16)f0.x; a[1]  = (bf16)f0.y; a[2]  = (bf16)f0.z; a[3]  = (bf16)f0.w;
        a[4]  = (bf16)f1.x; a[5]  = (bf16)f1.y; a[6]  = (bf16)f1.z; a[7]  = (bf16)f1.w;
        a[8]  = (bf16)f2.x; a[9]  = (bf16)f2.y; a[10] = (bf16)f2.z; a[11] = (bf16)f2.w;
        a[12] = (bf16)f3.x; a[13] = (bf16)f3.y; a[14] = (bf16)f3.z; a[15] = (bf16)f3.w;

#pragma unroll
        for (int n = 0; n < 8; ++n) {
            // B fragment from LDS: 16 bf16 = two 16B ds_load_b128 halves
            const bf16* bp = sW + (n * 16 + l16) * WPITCH + k0 + kHalf * 8;
            const v8bf b0 = *(const v8bf*)(bp);
            const v8bf b1 = *(const v8bf*)(bp + 16);
            const v16bf b = __builtin_shufflevector(
                b0, b1, 0, 1, 2, 3, 4, 5, 6, 7, 8, 9, 10, 11, 12, 13, 14, 15);
            acc[n] = __builtin_amdgcn_wmma_f32_16x16x32_bf16(
                false, a, false, b, (short)0, acc[n], false, false);
        }
    }

    // ---- epilogue: + bias, guarded stores (f32)
#pragma unroll
    for (int n = 0; n < 8; ++n) {
        const int   col = n * 16 + l16;
        const float bv  = bias[col];
#pragma unroll
        for (int r = 0; r < 8; ++r) {
            const int mr = row0 + kHalf * 8 + r;
            if (mr < M) {
                const float v = acc[n][r] + bv;
                out[(long)mr * CDIM + col] = v;
                if (out2) out2[(long)mr * CDIM + col] = v;
            }
        }
    }
}

// ---------------------------------------------------------------------------
// agg = h  (float4 streaming copy; GIN (1+eps)*x term with eps=0)
// ---------------------------------------------------------------------------
__global__ void copy_f4_kernel(const float4* __restrict__ src,
                               float4* __restrict__ dst, long n) {
    long i = blockIdx.x * (long)blockDim.x + threadIdx.x;
    const long stride = (long)gridDim.x * blockDim.x;
    for (; i < n; i += stride) dst[i] = src[i];
}

// ---------------------------------------------------------------------------
// agg[dst] += h[src] : one wave per edge, 32 lanes x float4 gather,
// 4 f32 atomics per lane. h/agg are L2-resident (25.6MB each, 192MB L2).
// Next iteration's source row is prefetched (global_prefetch_b8) to overlap
// the index-load -> gather dependency chain.
// ---------------------------------------------------------------------------
__global__ void scatter_add_kernel(const float* __restrict__ h,
                                   const int* __restrict__ ei,
                                   float* __restrict__ agg) {
    const int lane = threadIdx.x & 31;
    long wave = (blockIdx.x * (long)blockDim.x + threadIdx.x) >> 5;
    const long nWaves = ((long)gridDim.x * blockDim.x) >> 5;
    for (long e = wave; e < N_EDGES; e += nWaves) {
        const long en = e + nWaves;
        if (en < N_EDGES) {
            const int sn = ei[en];
            __builtin_prefetch(h + (long)sn * CDIM + lane * 4, 0, 3);
        }
        const int s = ei[e];
        const int d = ei[N_EDGES + e];
        const float4 v = ((const float4*)(h + (long)s * CDIM))[lane];
        float* dp = agg + (long)d * CDIM + lane * 4;
        atomicAdd(dp + 0, v.x);
        atomicAdd(dp + 1, v.y);
        atomicAdd(dp + 2, v.z);
        atomicAdd(dp + 3, v.w);
    }
}

// ---------------------------------------------------------------------------
// BatchNorm (training mode, biased variance) in two passes + fused ReLU.
// ---------------------------------------------------------------------------
__global__ void zero_stats_kernel(float* __restrict__ stats) {
    stats[threadIdx.x] = 0.f;                      // 256 threads: sums + sumsq
}

__global__ void stats_kernel(const float* __restrict__ t,
                             float* __restrict__ stats) {
    const int c = threadIdx.x;                     // 128 channels
    float s = 0.f, s2 = 0.f;
    for (int r = blockIdx.x; r < N_NODES; r += gridDim.x) {
        const float v = t[(long)r * CDIM + c];     // coalesced per row
        s += v; s2 += v * v;
    }
    atomicAdd(&stats[c], s);
    atomicAdd(&stats[CDIM + c], s2);
}

__global__ void bn_relu_kernel(const float* __restrict__ t,
                               const float* __restrict__ stats,
                               const float* __restrict__ gamma,
                               const float* __restrict__ beta,
                               float* __restrict__ outA,
                               float* __restrict__ outB) {
    const float invN = 1.0f / (float)N_NODES;
    const long total = (long)N_NODES * CDIM;
    long i = blockIdx.x * (long)blockDim.x + threadIdx.x;
    const long stride = (long)gridDim.x * blockDim.x;
    for (; i < total; i += stride) {
        const int c = (int)(i & (CDIM - 1));
        const float mu  = stats[c] * invN;
        const float var = stats[CDIM + c] * invN - mu * mu;
        float v = (t[i] - mu) * rsqrtf(var + BN_EPS) * gamma[c] + beta[c];
        v = v > 0.f ? v : 0.f;
        outA[i] = v;
        if (outB) outB[i] = v;
    }
}

// ---------------------------------------------------------------------------
// Driver. Inputs (setup_inputs order):
//  0:x 1:edge_index 2:batch 3:fh_W 4:fh_b 5:W1 6:b1 7:g1 8:be1 9:W2 10:b2 11:g2 12:be2
// Output: [4, 50, 1000, 128] f32 = 4 contiguous [50000*128] snapshots
// (to_dense_batch is a pure reshape for equal-size graphs).
// ---------------------------------------------------------------------------
extern "C" void kernel_launch(void* const* d_in, const int* in_sizes, int n_in,
                              void* d_out, int out_size, void* d_ws, size_t ws_size,
                              hipStream_t stream) {
    (void)in_sizes; (void)n_in; (void)out_size; (void)ws_size;
    const float* x    = (const float*)d_in[0];
    const int*   ei   = (const int*)d_in[1];       // [2, N_EDGES], src then dst
    const float* fh_W = (const float*)d_in[3];
    const float* fh_b = (const float*)d_in[4];
    const float* W1   = (const float*)d_in[5];
    const float* b1   = (const float*)d_in[6];
    const float* g1   = (const float*)d_in[7];
    const float* be1  = (const float*)d_in[8];
    const float* W2   = (const float*)d_in[9];
    const float* b2   = (const float*)d_in[10];
    const float* g2   = (const float*)d_in[11];
    const float* be2  = (const float*)d_in[12];
    float* out = (float*)d_out;

    const long NC = (long)N_NODES * CDIM;          // 6.4M floats
    float* hbuf   = (float*)d_ws;                  // current node features
    float* aggbuf = hbuf + NC;                     // h + neighbor sum
    float* tbuf   = aggbuf + NC;                   // pre-BN linear output
    bf16*  wbf    = (bf16*)(tbuf + NC);            // 7 x 128x128 bf16 (col-major)
    float* stats  = (float*)(wbf + 7 * CDIM * CDIM); // 256 f32: sum | sumsq

    // --- convert weights to bf16 col-major once per call
    prep_w_kernel<<<CDIM, CDIM, 0, stream>>>(fh_W, wbf);
    for (int l = 0; l < 3; ++l) {
        prep_w_kernel<<<CDIM, CDIM, 0, stream>>>(W1 + (long)l * CDIM * CDIM,
                                                 wbf + (long)(1 + l) * CDIM * CDIM);
        prep_w_kernel<<<CDIM, CDIM, 0, stream>>>(W2 + (long)l * CDIM * CDIM,
                                                 wbf + (long)(4 + l) * CDIM * CDIM);
    }

    const int gemmBlocks = (N_NODES + 63) / 64;    // 782 blocks x 4 waves

    // --- first head: h = x @ fh_W + fh_b ; snapshot 0
    gemm_bias_kernel<<<gemmBlocks, 128, 0, stream>>>(x, wbf, fh_b, hbuf, out, N_NODES);

    for (int l = 0; l < 3; ++l) {
        // GIN aggregate: agg = h + sum_{j->i} h_j
        copy_f4_kernel<<<4096, 256, 0, stream>>>((const float4*)hbuf,
                                                 (float4*)aggbuf, NC / 4);
        scatter_add_kernel<<<2048, 256, 0, stream>>>(hbuf, ei, aggbuf);

        // MLP layer 1: t = agg @ W1[l] + b1[l]; BN+ReLU -> h
        gemm_bias_kernel<<<gemmBlocks, 128, 0, stream>>>(
            aggbuf, wbf + (long)(1 + l) * CDIM * CDIM, b1 + (long)l * CDIM,
            tbuf, nullptr, N_NODES);
        zero_stats_kernel<<<1, 256, 0, stream>>>(stats);
        stats_kernel<<<512, 128, 0, stream>>>(tbuf, stats);
        bn_relu_kernel<<<4096, 256, 0, stream>>>(tbuf, stats,
            g1 + (long)l * CDIM, be1 + (long)l * CDIM, hbuf, nullptr);

        // MLP layer 2: t = h @ W2[l] + b2[l]; BN+ReLU -> h (+ snapshot l+1)
        gemm_bias_kernel<<<gemmBlocks, 128, 0, stream>>>(
            hbuf, wbf + (long)(4 + l) * CDIM * CDIM, b2 + (long)l * CDIM,
            tbuf, nullptr, N_NODES);
        zero_stats_kernel<<<1, 256, 0, stream>>>(stats);
        stats_kernel<<<512, 128, 0, stream>>>(tbuf, stats);
        bn_relu_kernel<<<4096, 256, 0, stream>>>(tbuf, stats,
            g2 + (long)l * CDIM, be2 + (long)l * CDIM, hbuf,
            out + (long)(l + 1) * NC);
    }
}
